// MalwareGNN_65481071395054
// MI455X (gfx1250) — matrix operand order repaired and checked by reading, not compiled, not run on hardware
//
#include <hip/hip_runtime.h>

typedef __bf16 bf16;
typedef __attribute__((ext_vector_type(16))) __bf16 v16bf;
typedef __attribute__((ext_vector_type(8)))  __bf16 v8bf;
typedef __attribute__((ext_vector_type(8)))  float  v8f;

#define HID 64
#define NFEAT 128

__device__ __forceinline__ void atomAddF(float* p, float v) {
    // native global_atomic_add_f32 (L2 atomic units), not a CAS loop
    unsafeAtomicAdd(p, v);
}

// ---------------- small utility kernels ----------------

__global__ void fill_f32(float* __restrict__ p, float v, int n) {
    int i = blockIdx.x * blockDim.x + threadIdx.x;
    if (i < n) p[i] = v;
}

__global__ void deg_edges(const int* __restrict__ dst, float* __restrict__ deg, int E) {
    int e = blockIdx.x * blockDim.x + threadIdx.x;
    if (e < E) atomAddF(&deg[dst[e]], 1.0f);
}

__global__ void make_dinv(const float* __restrict__ deg, float* __restrict__ dinv, int n) {
    int i = blockIdx.x * blockDim.x + threadIdx.x;
    if (i < n) dinv[i] = rsqrtf(fmaxf(deg[i], 1.0f));
}

__global__ void cvt_bf16(const float* __restrict__ x, bf16* __restrict__ y, int n) {
    int i = blockIdx.x * blockDim.x + threadIdx.x;
    if (i < n) y[i] = (bf16)x[i];
}

// Pre-swizzle W (K x 64, row-major f32) into WMMA B-fragment layout (bf16):
// fragment f = ks*4 + nt holds the 32x16 tile (K rows ks*32.., N cols nt*16..)
// laid out lane-major per ISA 7.12.2: lane = ((k%32)/8%2)*16 + n%16, elem = k%8 + (k%32)/16*8.
__global__ void prep_w(const float* __restrict__ W, bf16* __restrict__ frag, int K) {
    int i = blockIdx.x * blockDim.x + threadIdx.x;
    if (i >= K * HID) return;
    int k = i >> 6, n = i & 63;
    int ks = k >> 5, kr = k & 31;
    int half = (kr >> 3) & 1, e = (kr & 7) + ((kr >> 4) << 3);
    int nt = n >> 4, nl = n & 15;
    int lane = half * 16 + nl;
    frag[(size_t)(ks * 4 + nt) * 512 + lane * 16 + e] = (bf16)W[(size_t)k * HID + n];
}

// ---------------- WMMA GEMM: H[N x 64] = A[N x K](bf16) @ Wfrag ----------------
// One wave per 16-row tile; 4 accumulators cover the full 64 output columns.
template <int K>
__global__ void gemm_wmma(const bf16* __restrict__ A, const bf16* __restrict__ Wfrag,
                          float* __restrict__ H, int n) {
    int wave = blockIdx.x * (blockDim.x >> 5) + (threadIdx.x >> 5);
    int lane = threadIdx.x & 31;
    int half = lane >> 4;
    int nl   = lane & 15;
    int nblk = (n + 15) >> 4;
    if (wave >= nblk) return;  // wave-uniform: EXEC stays all-ones for WMMA

    int arow = wave * 16 + nl;
    if (arow > n - 1) arow = n - 1;  // clamp tail loads (stores are guarded)
    const bf16* Ap = A + (size_t)arow * K;

    v8f acc0 = {}, acc1 = {}, acc2 = {}, acc3 = {};

#pragma unroll
    for (int ks = 0; ks < K / 32; ++ks) {
        // A fragment (16x32 bf16): per 7.12.2, lane holds row M=lane%16;
        // elems 0..7 -> K = base..base+7, elems 8..15 -> K = base+16..base+23,
        // base = ks*32 + (lane>=16 ? 8 : 0).
        const bf16* ap = Ap + ks * 32 + half * 8;
        v8bf lo = *(const v8bf*)(ap);
        v8bf hi = *(const v8bf*)(ap + 16);
        v16bf a;
#pragma unroll
        for (int i = 0; i < 8; ++i) { a[i] = lo[i]; a[i + 8] = hi[i]; }

        const bf16* bp = Wfrag + (size_t)(ks * 4) * 512 + lane * 16;
        v16bf b0 = *(const v16bf*)(bp);
        v16bf b1 = *(const v16bf*)(bp + 512);
        v16bf b2 = *(const v16bf*)(bp + 1024);
        v16bf b3 = *(const v16bf*)(bp + 1536);

        acc0 = __builtin_amdgcn_wmma_f32_16x16x32_bf16(false, a, false, b0, (short)0, acc0, false, false);
        acc1 = __builtin_amdgcn_wmma_f32_16x16x32_bf16(false, a, false, b1, (short)0, acc1, false, false);
        acc2 = __builtin_amdgcn_wmma_f32_16x16x32_bf16(false, a, false, b2, (short)0, acc2, false, false);
        acc3 = __builtin_amdgcn_wmma_f32_16x16x32_bf16(false, a, false, b3, (short)0, acc3, false, false);
    }

    // C/D layout: vgpr r, lanes 0-15 -> M=r, lanes 16-31 -> M=8+r, N=lane%16
    int rbase = wave * 16 + half * 8;
#pragma unroll
    for (int r = 0; r < 8; ++r) {
        int row = rbase + r;
        if (row < n) {
            float* o = H + (size_t)row * HID + nl;
            o[0]  = acc0[r];
            o[16] = acc1[r];
            o[32] = acc2[r];
            o[48] = acc3[r];
        }
    }
}

// ---------------- aggregation ----------------

// out[n] = H[n] * dinv[n]^2   (self-loop term; also initializes out, no memset needed)
__global__ void selfloop_init(const float* __restrict__ H, const float* __restrict__ dinv,
                              float* __restrict__ out, int n) {
    int i = blockIdx.x * blockDim.x + threadIdx.x;
    if (i >= n * 16) return;
    int node = i >> 4, c = i & 15;
    float d = dinv[node];
    float s = d * d;
    const float4 v = *(const float4*)(H + (size_t)node * HID + c * 4);
    float4 o = make_float4(v.x * s, v.y * s, v.z * s, v.w * s);
    *(float4*)(out + (size_t)node * HID + c * 4) = o;
}

// out[dst] += H[src] * dinv[src]*dinv[dst]  (one thread per edge x 4-feature chunk)
__global__ void edge_scatter(const float* __restrict__ H, const int* __restrict__ src,
                             const int* __restrict__ dst, const float* __restrict__ dinv,
                             float* __restrict__ out, int E) {
    long long i = (long long)blockIdx.x * blockDim.x + threadIdx.x;
    if (i >= (long long)E * 16) return;
    int e = (int)(i >> 4), c = (int)(i & 15);
    int s = src[e], d = dst[e];
    float nrm = dinv[s] * dinv[d];
    const float4 v = *(const float4*)(H + (size_t)s * HID + c * 4);
    float* o = out + (size_t)d * HID + c * 4;
    atomAddF(o + 0, v.x * nrm);
    atomAddF(o + 1, v.y * nrm);
    atomAddF(o + 2, v.z * nrm);
    atomAddF(o + 3, v.w * nrm);
}

// y = bf16(relu(agg + b))  -> feeds next layer's WMMA A operand
__global__ void post_relu_bf16(const float* __restrict__ agg, const float* __restrict__ b,
                               bf16* __restrict__ y, int n) {
    int i = blockIdx.x * blockDim.x + threadIdx.x;
    if (i >= n * 16) return;
    int node = i >> 4, c = i & 15;
    const float4 v  = *(const float4*)(agg + (size_t)node * HID + c * 4);
    const float4 bb = *(const float4*)(b + c * 4);
    bf16* o = y + (size_t)node * HID + c * 4;
    o[0] = (bf16)fmaxf(v.x + bb.x, 0.0f);
    o[1] = (bf16)fmaxf(v.y + bb.y, 0.0f);
    o[2] = (bf16)fmaxf(v.z + bb.z, 0.0f);
    o[3] = (bf16)fmaxf(v.w + bb.w, 0.0f);
}

// summ[batch[n]] += agg[n] + b3 ; cnt[batch[n]] += 1
__global__ void pool_accum(const float* __restrict__ agg, const float* __restrict__ b3,
                           const int* __restrict__ batch, float* __restrict__ summ,
                           float* __restrict__ cnt, int n) {
    int i = blockIdx.x * blockDim.x + threadIdx.x;
    if (i >= n * 16) return;
    int node = i >> 4, c = i & 15;
    int g = batch[node];
    const float4 v  = *(const float4*)(agg + (size_t)node * HID + c * 4);
    const float4 bb = *(const float4*)(b3 + c * 4);
    float* o = summ + (size_t)g * HID + c * 4;
    atomAddF(o + 0, v.x + bb.x);
    atomAddF(o + 1, v.y + bb.y);
    atomAddF(o + 2, v.z + bb.z);
    atomAddF(o + 3, v.w + bb.w);
    if (c == 0) atomAddF(&cnt[g], 1.0f);
}

// ---------------- head: one wave per graph ----------------
// lanes 0..15 = classes; min over 3 centroids of squared distance; shfl min for reject
__global__ void head_kernel(const float* __restrict__ summ, const float* __restrict__ cnt,
                            const float* __restrict__ cent, const float* __restrict__ rbias,
                            float* __restrict__ out) {
    int g = blockIdx.x;
    int lane = threadIdx.x;
    __shared__ float emb[HID];
    float inv = 1.0f / fmaxf(cnt[g], 1.0f);
    for (int f = lane; f < HID; f += 32) emb[f] = summ[(size_t)g * HID + f] * inv;
    __syncthreads();

    float dpc = 3.4e38f;
    if (lane < 16) {
#pragma unroll
        for (int j = 0; j < 3; ++j) {
            const float* cp = cent + (size_t)(lane * 3 + j) * HID;
            float dist = 0.0f;
#pragma unroll
            for (int f = 0; f < HID; ++f) {
                float t = emb[f] - cp[f];
                dist = fmaf(t, t, dist);
            }
            dpc = fminf(dpc, dist);
        }
        out[(size_t)g * 17 + lane] = -dpc;
    }
    // min over lanes 0..15 (xor 8/4/2/1 keeps the two half-groups separate)
    for (int off = 8; off > 0; off >>= 1) dpc = fminf(dpc, __shfl_xor(dpc, off, 32));
    if (lane == 0) out[(size_t)g * 17 + 16] = dpc - rbias[0];
}

// ---------------- launcher ----------------

extern "C" void kernel_launch(void* const* d_in, const int* in_sizes, int n_in,
                              void* d_out, int out_size, void* d_ws, size_t ws_size,
                              hipStream_t stream) {
    const float* x     = (const float*)d_in[0];
    const int*   ei    = (const int*)d_in[1];
    const int*   batch = (const int*)d_in[2];
    const float* W1 = (const float*)d_in[3];
    const float* b1 = (const float*)d_in[4];
    const float* W2 = (const float*)d_in[5];
    const float* b2 = (const float*)d_in[6];
    const float* W3 = (const float*)d_in[7];
    const float* b3 = (const float*)d_in[8];
    const float* cent  = (const float*)d_in[9];
    const float* rbias = (const float*)d_in[10];

    const int N = in_sizes[0] / NFEAT;
    const int E = in_sizes[1] / 2;
    const int G = out_size / 17;
    const int* src = ei;
    const int* dst = ei + E;

    // workspace carve-out (256B aligned)
    char* base = (char*)d_ws;
    size_t off = 0;
    auto alloc = [&](size_t bytes) -> void* {
        void* p = base + off;
        off = (off + bytes + 255) & ~(size_t)255;
        return p;
    };
    bf16*  Xb   = (bf16*)alloc((size_t)N * NFEAT * sizeof(bf16));
    float* Hbuf = (float*)alloc((size_t)N * HID * sizeof(float));
    float* Agg  = (float*)alloc((size_t)N * HID * sizeof(float));
    bf16*  Hb   = (bf16*)alloc((size_t)N * HID * sizeof(bf16));
    float* deg  = (float*)alloc((size_t)N * sizeof(float));
    float* dinv = (float*)alloc((size_t)N * sizeof(float));
    bf16*  F1   = (bf16*)alloc((size_t)4 * 4 * 512 * sizeof(bf16));  // K=128
    bf16*  F2   = (bf16*)alloc((size_t)2 * 4 * 512 * sizeof(bf16));  // K=64
    bf16*  F3   = (bf16*)alloc((size_t)2 * 4 * 512 * sizeof(bf16));  // K=64
    float* summ = (float*)alloc((size_t)G * HID * sizeof(float));
    float* cnt  = (float*)alloc((size_t)G * sizeof(float));
    (void)ws_size; (void)n_in;

    const int T = 256;
    auto cdiv = [](long long a, long long b) { return (int)((a + b - 1) / b); };

    // degree / normalization
    fill_f32<<<cdiv(N, T), T, 0, stream>>>(deg, 1.0f, N);           // self-loop
    fill_f32<<<cdiv(G * HID, T), T, 0, stream>>>(summ, 0.0f, G * HID);
    fill_f32<<<cdiv(G, T), T, 0, stream>>>(cnt, 0.0f, G);
    deg_edges<<<cdiv(E, T), T, 0, stream>>>(dst, deg, E);
    make_dinv<<<cdiv(N, T), T, 0, stream>>>(deg, dinv, N);

    // inputs -> bf16 ; weights -> fragment layout
    cvt_bf16<<<cdiv((long long)N * NFEAT, T), T, 0, stream>>>(x, Xb, N * NFEAT);
    prep_w<<<cdiv(NFEAT * HID, T), T, 0, stream>>>(W1, F1, NFEAT);
    prep_w<<<cdiv(HID * HID, T), T, 0, stream>>>(W2, F2, HID);
    prep_w<<<cdiv(HID * HID, T), T, 0, stream>>>(W3, F3, HID);

    const int nblk = (N + 15) / 16;           // 16-row tiles
    const int gemmGrid = cdiv(nblk, T / 32);  // 8 waves per block
    const int nodeGrid = cdiv((long long)N * 16, T);
    const int edgeGrid = cdiv((long long)E * 16, T);

    // layer 1
    gemm_wmma<NFEAT><<<gemmGrid, T, 0, stream>>>(Xb, F1, Hbuf, N);
    selfloop_init<<<nodeGrid, T, 0, stream>>>(Hbuf, dinv, Agg, N);
    edge_scatter<<<edgeGrid, T, 0, stream>>>(Hbuf, src, dst, dinv, Agg, E);
    post_relu_bf16<<<nodeGrid, T, 0, stream>>>(Agg, b1, Hb, N);

    // layer 2
    gemm_wmma<HID><<<gemmGrid, T, 0, stream>>>(Hb, F2, Hbuf, N);
    selfloop_init<<<nodeGrid, T, 0, stream>>>(Hbuf, dinv, Agg, N);
    edge_scatter<<<edgeGrid, T, 0, stream>>>(Hbuf, src, dst, dinv, Agg, E);
    post_relu_bf16<<<nodeGrid, T, 0, stream>>>(Agg, b2, Hb, N);

    // layer 3 (no relu; bias folded into pooling)
    gemm_wmma<HID><<<gemmGrid, T, 0, stream>>>(Hb, F3, Hbuf, N);
    selfloop_init<<<nodeGrid, T, 0, stream>>>(Hbuf, dinv, Agg, N);
    edge_scatter<<<edgeGrid, T, 0, stream>>>(Hbuf, src, dst, dinv, Agg, E);
    pool_accum<<<nodeGrid, T, 0, stream>>>(Agg, b3, batch, summ, cnt, N);

    // head
    head_kernel<<<G, 32, 0, stream>>>(summ, cnt, cent, rbias, (float*)d_out);
}
